// GPTOSSExperts_18580028523159
// MI455X (gfx1250) — compile-verified, matrix operand order checked by reading, not run on hardware
//
#include <hip/hip_runtime.h>

// ---------------------------------------------------------------------------
// GPT-OSS MoE experts on MI455X (gfx1250, wave32, WMMA).
// Routed-only compute (103 GFLOP vs 412 dense). bf16 WMMA + f32 accum keeps
// compute (~40us) near the 768MB / 23.3TB/s = 33us weight-stream floor.
// Block tile: M=32 tokens x N=128 cols, K-chunk 64; 8 waves, each wave owns a
// 16x32 output tile (2 accumulators, 4 v_wmma per barrier round).
// Down-GEMM A tile staged with GLOBAL_LOAD_ASYNC_TO_LDS_B128 (ASYNCcnt).
// ---------------------------------------------------------------------------

typedef __attribute__((ext_vector_type(16))) __bf16 v16bf;
typedef __attribute__((ext_vector_type(8)))  float  v8f;

constexpr int B_TOK   = 1024;
constexpr int TOPK    = 4;
constexpr int NEXP    = 16;
constexpr int H_DIM   = 2048;
constexpr int I_DIM   = 2048;
constexpr int TWO_I   = 4096;
constexpr int NASSIGN = B_TOK * TOPK;   // 4096 (token, k) assignments

constexpr float SWIGLU_LIMIT = 7.0f;
constexpr float ALPHA        = 1.702f;

__device__ __forceinline__ unsigned short f32_to_bf16(float f) {
  unsigned u = __builtin_bit_cast(unsigned, f);
  u += 0x7fffu + ((u >> 16) & 1u);           // round-to-nearest-even
  return (unsigned short)(u >> 16);
}

// RNE pack of two f32 into packed bf16x2 via one v_perm_b32 (bytes {7,6,3,2}).
__device__ __forceinline__ unsigned pack_bf16x2(float lo, float hi) {
  unsigned ul = __builtin_bit_cast(unsigned, lo);
  unsigned uh = __builtin_bit_cast(unsigned, hi);
  ul += 0x7fffu + ((ul >> 16) & 1u);
  uh += 0x7fffu + ((uh >> 16) & 1u);
  return __builtin_amdgcn_perm(uh, ul, 0x07060302u);
}

// Async global->LDS 16B copy (per lane), tracked by ASYNCcnt.
// lds_off: LDS byte offset (generic shared pointer truncated to 32 bits ==
// LDS offset per the gfx1250 aperture layout).
__device__ __forceinline__ void async_copy_b128(unsigned lds_off,
                                                const void* gptr) {
  asm volatile("global_load_async_to_lds_b128 %0, %1, off"
               :: "v"(lds_off), "v"((unsigned long long)gptr)
               : "memory");
}
__device__ __forceinline__ void wait_asynccnt0() {
  asm volatile("s_wait_asynccnt 0x0" ::: "memory");
}

// ---------------------------------------------------------------------------
// Routing: per-expert token lists via atomic scatter.
// ---------------------------------------------------------------------------
__global__ void k_zero_counters(int* cnt) {
  if (threadIdx.x < NEXP) cnt[threadIdx.x] = 0;
}

__global__ void k_route(const int* __restrict__ idx, int* cnt, int* slots) {
  int a = blockIdx.x * blockDim.x + threadIdx.x;
  if (a < NASSIGN) {
    int e = idx[a];
    int p = atomicAdd(&cnt[e], 1);
    slots[e * NASSIGN + p] = a;   // assignment id: a = b*TOPK + k
  }
}

// 16-bit A/B fragment K pattern (CDNA5 ISA 7.12.2): lane group
// (lane>=16) selects K+8; VGPR j holds packed pair {2j | 16+2(j-4)}.
__device__ __forceinline__ int frag_k_of(int j, int lane) {
  int kg = (lane >= 16) ? 8 : 0;
  return kg + ((j < 4) ? (2 * j) : (16 + 2 * (j - 4)));
}

// Load a 16x32 bf16 fragment from LDS tile (row stride 64 bf16).
__device__ __forceinline__ v16bf load_frag(const unsigned short* lds,
                                           int row, int kc, int lane) {
  union { v16bf v; unsigned u[8]; } f;
#pragma unroll
  for (int j = 0; j < 8; ++j) {
    int kk = kc + frag_k_of(j, lane);
    f.u[j] = *(const unsigned*)&lds[row * 64 + kk];
  }
  return f.v;
}

// ---------------------------------------------------------------------------
// Kernel 1: h = swiglu(t @ gate_up[e]^T + bias)  ->  hbuf (bf16)
// ---------------------------------------------------------------------------
__global__ __launch_bounds__(256)
void k_gateup_swiglu(const float* __restrict__ t,
                     const float* __restrict__ w,       // (E, 2I, H)
                     const float* __restrict__ bias,    // (E, 2I)
                     const int*   __restrict__ cnt,
                     const int*   __restrict__ slots,
                     unsigned short* __restrict__ hbuf) // (NASSIGN, I) bf16
{
  const int e     = blockIdx.y;
  const int cBase = blockIdx.x * 128;         // column tile in [0, 2I)
  const int n     = cnt[e];
  if (n == 0) return;

  const int tid  = threadIdx.x;
  const int lane = tid & 31;
  const int wid  = tid >> 5;
  const int wm   = wid >> 2;                  // 0..1
  const int wn   = wid & 3;                   // 0..3

  __shared__ unsigned short lA[32 * 64];      // tokens x K
  __shared__ unsigned short lB[128 * 64];     // cols   x K
  __shared__ float          lC[32 * 128];
  __shared__ int            rowA[32];

  const float* wE = w + ((size_t)e * TWO_I + cBase) * H_DIM;
  const int nTiles = (n + 31) >> 5;

  // Per-thread staging geometry (hoisted): 16 float4 per 64-wide row.
  const int sR = tid >> 4;                    // 0..15
  const int sC = (tid & 15) * 4;              // float offset within K-chunk

  for (int tt = 0; tt < nTiles; ++tt) {
    if (tid < 32) {
      int p = tt * 32 + tid;
      rowA[tid] = slots[e * NASSIGN + min(p, n - 1)];  // clamp: branchless gather
    }
    __syncthreads();

    // Hoisted token-row base pointers for A staging (rows sR and sR+16).
    const float* aSrc0 = t + (size_t)(rowA[sR]      >> 2) * H_DIM + sC;
    const float* aSrc1 = t + (size_t)(rowA[sR + 16] >> 2) * H_DIM + sC;

    v8f acc0 = {}, acc1 = {};
    for (int k0 = 0; k0 < H_DIM; k0 += 64) {
      // ---- stage A: 32x64 fp32 -> bf16, two float4 per thread ----
      {
        float4 v0 = *(const float4*)(aSrc0 + k0);
        float4 v1 = *(const float4*)(aSrc1 + k0);
        *(uint2*)&lA[sR * 64 + sC]        = make_uint2(pack_bf16x2(v0.x, v0.y),
                                                       pack_bf16x2(v0.z, v0.w));
        *(uint2*)&lA[(sR + 16) * 64 + sC] = make_uint2(pack_bf16x2(v1.x, v1.y),
                                                       pack_bf16x2(v1.z, v1.w));
      }
      // ---- stage B: 128x64 fp32 -> bf16, eight float4 per thread ----
#pragma unroll
      for (int it = 0; it < 8; ++it) {
        int r = sR + it * 16;
        float4 v = *(const float4*)(wE + (size_t)r * H_DIM + k0 + sC);
        *(uint2*)&lB[r * 64 + sC] = make_uint2(pack_bf16x2(v.x, v.y),
                                               pack_bf16x2(v.z, v.w));
      }
      if (k0 + 64 < H_DIM)  // pull next weight K-slab toward L2
        __builtin_prefetch(wE + (size_t)sR * H_DIM + k0 + 64 + sC, 0, 1);
      __syncthreads();

      const int mA  = wm * 16 + (lane & 15);
      const int nB0 = wn * 32 + (lane & 15);
      v16bf a0  = load_frag(lA, mA, 0, lane);
      v16bf a1  = load_frag(lA, mA, 32, lane);
      v16bf b00 = load_frag(lB, nB0, 0, lane);
      v16bf b01 = load_frag(lB, nB0, 32, lane);
      v16bf b10 = load_frag(lB, nB0 + 16, 0, lane);
      v16bf b11 = load_frag(lB, nB0 + 16, 32, lane);
      acc0 = __builtin_amdgcn_wmma_f32_16x16x32_bf16(false, a0, false, b00, (short)0, acc0, false, false);
      acc0 = __builtin_amdgcn_wmma_f32_16x16x32_bf16(false, a1, false, b01, (short)0, acc0, false, false);
      acc1 = __builtin_amdgcn_wmma_f32_16x16x32_bf16(false, a0, false, b10, (short)0, acc1, false, false);
      acc1 = __builtin_amdgcn_wmma_f32_16x16x32_bf16(false, a1, false, b11, (short)0, acc1, false, false);
      __syncthreads();
    }

    // C layout: vgpr r, lane l -> M = r + 8*(l>=16), N = l&15
#pragma unroll
    for (int r = 0; r < 8; ++r) {
      int mm = wm * 16 + r + ((lane >= 16) ? 8 : 0);
      lC[mm * 128 + wn * 32 + (lane & 15)]      = acc0[r];
      lC[mm * 128 + wn * 32 + 16 + (lane & 15)] = acc1[r];
    }
    __syncthreads();

    // bias + swiglu on (even, odd) column pairs -> bf16 hbuf (64 outs/row)
#pragma unroll
    for (int it = 0; it < 8; ++it) {
      int i = tid + it * 256;
      int r = i >> 6, oi = i & 63;
      if (tt * 32 + r < n) {
        int a  = rowA[r];
        int c0 = cBase + 2 * oi;
        float g = lC[r * 128 + 2 * oi]     + bias[(size_t)e * TWO_I + c0];
        float u = lC[r * 128 + 2 * oi + 1] + bias[(size_t)e * TWO_I + c0 + 1];
        g = fminf(g, SWIGLU_LIMIT);
        u = fminf(fmaxf(u, -SWIGLU_LIMIT), SWIGLU_LIMIT);
        float s   = 1.0f / (1.0f + __expf(-ALPHA * g));
        float act = g * s * (u + 1.0f);
        hbuf[(size_t)a * I_DIM + (cBase >> 1) + oi] = f32_to_bf16(act);
      }
    }
    __syncthreads();
  }
}

// ---------------------------------------------------------------------------
// Kernel 2: out[a, :] = h[a] @ down[e]^T + down_bias[e]   (out fp32)
// A tile (already bf16) staged via async global->LDS copies.
// ---------------------------------------------------------------------------
__global__ __launch_bounds__(256)
void k_down(const unsigned short* __restrict__ hbuf,  // (NASSIGN, I) bf16
            const float* __restrict__ w,              // (E, H, I)
            const float* __restrict__ bias,           // (E, H)
            const int*   __restrict__ cnt,
            const int*   __restrict__ slots,
            float* __restrict__ out)                  // (B, K, H) fp32
{
  const int e     = blockIdx.y;
  const int nBase = blockIdx.x * 128;         // column tile in [0, H)
  const int n     = cnt[e];
  if (n == 0) return;

  const int tid  = threadIdx.x;
  const int lane = tid & 31;
  const int wid  = tid >> 5;
  const int wm   = wid >> 2;
  const int wn   = wid & 3;

  __shared__ unsigned short lA[32 * 64];
  __shared__ unsigned short lB[128 * 64];
  __shared__ float          lC[32 * 128];
  __shared__ int            rowA[32];

  const float* wE = w + ((size_t)e * H_DIM + nBase) * I_DIM;
  const int nTiles = (n + 31) >> 5;

  const int sR = tid >> 4;                    // B staging: 16 float4 per row
  const int sC = (tid & 15) * 4;
  const int aR = tid >> 3;                    // A staging: 8 x 16B per row
  const int aC = (tid & 7) * 8;               // bf16-element offset

  // LDS byte offset of this thread's A-tile slot (generic ptr low 32 bits).
  const unsigned ldsA_off = (unsigned)(unsigned long long)&lA[aR * 64 + aC];

  for (int tt = 0; tt < nTiles; ++tt) {
    if (tid < 32) {
      int p = tt * 32 + tid;
      rowA[tid] = slots[e * NASSIGN + min(p, n - 1)];
    }
    __syncthreads();

    const unsigned short* aSrc = hbuf + (size_t)rowA[aR] * I_DIM + aC;

    v8f acc0 = {}, acc1 = {};
    for (int k0 = 0; k0 < I_DIM; k0 += 64) {
      // ---- stage A: 32x64 bf16 via async global->LDS (16B per lane) ----
      async_copy_b128(ldsA_off, aSrc + k0);
      // ---- stage B: 128x64 fp32 -> bf16, eight float4 per thread ----
#pragma unroll
      for (int it = 0; it < 8; ++it) {
        int r = sR + it * 16;
        float4 v = *(const float4*)(wE + (size_t)r * I_DIM + k0 + sC);
        *(uint2*)&lB[r * 64 + sC] = make_uint2(pack_bf16x2(v.x, v.y),
                                               pack_bf16x2(v.z, v.w));
      }
      if (k0 + 64 < I_DIM)
        __builtin_prefetch(wE + (size_t)sR * I_DIM + k0 + 64 + sC, 0, 1);
      wait_asynccnt0();                       // A tile landed in LDS
      __syncthreads();

      const int mA  = wm * 16 + (lane & 15);
      const int nB0 = wn * 32 + (lane & 15);
      v16bf a0  = load_frag(lA, mA, 0, lane);
      v16bf a1  = load_frag(lA, mA, 32, lane);
      v16bf b00 = load_frag(lB, nB0, 0, lane);
      v16bf b01 = load_frag(lB, nB0, 32, lane);
      v16bf b10 = load_frag(lB, nB0 + 16, 0, lane);
      v16bf b11 = load_frag(lB, nB0 + 16, 32, lane);
      acc0 = __builtin_amdgcn_wmma_f32_16x16x32_bf16(false, a0, false, b00, (short)0, acc0, false, false);
      acc0 = __builtin_amdgcn_wmma_f32_16x16x32_bf16(false, a1, false, b01, (short)0, acc0, false, false);
      acc1 = __builtin_amdgcn_wmma_f32_16x16x32_bf16(false, a0, false, b10, (short)0, acc1, false, false);
      acc1 = __builtin_amdgcn_wmma_f32_16x16x32_bf16(false, a1, false, b11, (short)0, acc1, false, false);
      __syncthreads();
    }

#pragma unroll
    for (int r = 0; r < 8; ++r) {
      int mm = wm * 16 + r + ((lane >= 16) ? 8 : 0);
      lC[mm * 128 + wn * 32 + (lane & 15)]      = acc0[r];
      lC[mm * 128 + wn * 32 + 16 + (lane & 15)] = acc1[r];
    }
    __syncthreads();

    // epilogue: bias add, scatter rows to out; 16 elems per thread
#pragma unroll
    for (int it = 0; it < 16; ++it) {
      int i = tid + it * 256;
      int r = i >> 7, cc = i & 127;
      if (tt * 32 + r < n) {
        int a = rowA[r];
        out[(size_t)a * H_DIM + nBase + cc] =
            lC[r * 128 + cc] + bias[(size_t)e * H_DIM + nBase + cc];
      }
    }
    __syncthreads();
  }
}

// ---------------------------------------------------------------------------
// Launch. Workspace: [cnt 256B][slots 256KB][... hbuf @1MB: 16MB bf16]
// ---------------------------------------------------------------------------
extern "C" void kernel_launch(void* const* d_in, const int* in_sizes, int n_in,
                              void* d_out, int out_size, void* d_ws, size_t ws_size,
                              hipStream_t stream) {
  const float* t      = (const float*)d_in[0];
  const int*   idx    = (const int*)  d_in[1];
  const float* gup    = (const float*)d_in[2];
  const float* gup_b  = (const float*)d_in[3];
  const float* dwn    = (const float*)d_in[4];
  const float* dwn_b  = (const float*)d_in[5];
  float*       out    = (float*)d_out;

  int* cnt   = (int*)d_ws;
  int* slots = cnt + 64;
  unsigned short* hbuf = (unsigned short*)((char*)d_ws + (1u << 20));

  k_zero_counters<<<1, 32, 0, stream>>>(cnt);
  k_route<<<NASSIGN / 256, 256, 0, stream>>>(idx, cnt, slots);
  k_gateup_swiglu<<<dim3(TWO_I / 128, NEXP), 256, 0, stream>>>(
      t, gup, gup_b, cnt, slots, hbuf);
  k_down<<<dim3(H_DIM / 128, NEXP), 256, 0, stream>>>(
      hbuf, dwn, dwn_b, cnt, slots, out);
}
